// MultiHeadAttention_30434138259989
// MI455X (gfx1250) — matrix operand order repaired and checked by reading, not compiled, not run on hardware
//
#include <hip/hip_runtime.h>
#include <hip/hip_bf16.h>

// MultiHeadAttention forward for MI455X (gfx1250), wave32 + WMMA f16->f32.
//
// Pipeline (all on `stream`):
//   proj_kernel<0> : Qh = (q @ Wq^T + bq) * (1/sqrt(d))  -> f16 [N,H,L,d]
//   proj_kernel<0> : Kh = (k @ Wk^T + bk)                -> f16 [N,H,L,d]
//   proj_kernel<1> : Vt = (v @ Wv^T + bv)                -> f16 [N,H,d,L] (transposed)
//   attn_kernel    : flash attention, LDS-staged K/V     -> f16 [N,L,E]
//   oproj_kernel   : out = att @ Wo^T + bo               -> f32 [N,L,E]
//
// Workspace: 4 * (4096*1024) f16 = 32 MB.

#define EMB   1024
#define HEADS 16
#define HD    64
#define SEQ   2048
#define BATCH 2
#define MROWS (BATCH * SEQ)   // 4096
#define KSTEPS (EMB / 32)     // 32 k-steps of 32 for the projections

typedef __attribute__((ext_vector_type(16))) _Float16 v16h;
typedef __attribute__((ext_vector_type(8)))  _Float16 v8h;
typedef __attribute__((ext_vector_type(8)))  float    v8f;
typedef __attribute__((ext_vector_type(4)))  float    v4f;
typedef __attribute__((ext_vector_type(4)))  int      v4i;

__device__ __forceinline__ v8f wmma_f16(v16h a, v16h b, v8f c) {
  // D = A(16x32 f16) * B(32x16 f16) + C(16x16 f32)
  return __builtin_amdgcn_wmma_f32_16x16x32_f16(
      /*neg_a=*/false, a, /*neg_b=*/false, b,
      /*c_mod=*/(short)0, c, /*reuse_a=*/false, /*reuse_b=*/false);
}

// ---- 16-lane butterfly reductions via DPP (pure VALU, no LDS traffic) -----
// quad_perm xor1 = 0xB1, quad_perm xor2 = 0x4E, row_half_mirror = 0x141,
// row_mirror = 0x140. A DPP "row" is 16 lanes, matching the C/D half layout.
template <int CTRL>
__device__ __forceinline__ float dppmov(float x) {
  int r = __builtin_amdgcn_update_dpp(0, __float_as_int(x), CTRL, 0xF, 0xF, true);
  return __int_as_float(r);
}
__device__ __forceinline__ float redmax16(float x) {
  x = fmaxf(x, dppmov<0xB1>(x));
  x = fmaxf(x, dppmov<0x4E>(x));
  x = fmaxf(x, dppmov<0x141>(x));
  x = fmaxf(x, dppmov<0x140>(x));
  return x;
}
__device__ __forceinline__ float redsum16(float x) {
  x += dppmov<0xB1>(x);
  x += dppmov<0x4E>(x);
  x += dppmov<0x141>(x);
  x += dppmov<0x140>(x);
  return x;
}

// ---- async global -> LDS copy (CDNA5 ASYNCcnt path), with safe fallback ---
#if defined(__has_builtin)
#  if __has_builtin(__builtin_amdgcn_global_load_async_to_lds_b128)
#    define USE_ASYNC_LDS 1
#  endif
#endif
#ifndef USE_ASYNC_LDS
#  define USE_ASYNC_LDS 0
#endif

typedef __attribute__((address_space(1))) v4i gas_v4i;  // global AS pointer elt
typedef __attribute__((address_space(3))) v4i las_v4i;  // LDS AS pointer elt

__device__ __forceinline__ void copy16_g2l(_Float16* l, const _Float16* g) {
#if USE_ASYNC_LDS
  __builtin_amdgcn_global_load_async_to_lds_b128(
      (gas_v4i*)(uintptr_t)g, (las_v4i*)(uintptr_t)l, 0, 0);
#else
  *(uint4*)l = *(const uint4*)g;
#endif
}

__device__ __forceinline__ void wait_async_lds() {
#if USE_ASYNC_LDS
#  if __has_builtin(__builtin_amdgcn_s_wait_asynccnt)
  __builtin_amdgcn_s_wait_asynccnt(0);
#  else
  asm volatile("s_wait_asynccnt 0" ::: "memory");
#  endif
#endif
}

// ---------------------------------------------------------------------------
// Projection: out = (X @ W^T + b) * scale, f32 in, f16 out.
// Each wave computes a 16(M) x 64(N) tile; block = 4 waves stacked in M.
// grid = (MROWS/64, EMB/64), block = 128.
// TRANSPOSED=0: out[n][h][l][dd]   (for Q, K)
// TRANSPOSED=1: out[n][h][dd][l]   (for V -> B-fragment friendly in P@V)
// ---------------------------------------------------------------------------
template <int TRANSPOSED>
__global__ __launch_bounds__(128) void proj_kernel(
    const float* __restrict__ X, const float* __restrict__ W,
    const float* __restrict__ bias, _Float16* __restrict__ out, float scale) {
  const int lane = threadIdx.x & 31;
  const int wave = threadIdx.x >> 5;
  const int half = lane >> 4;   // which 16-lane half
  const int l16  = lane & 15;

  const int mBase = (blockIdx.x * 4 + wave) * 16;
  const int nBase = blockIdx.y * 64;

  v8f acc[4] = {};
  const float* xrow = X + (size_t)(mBase + l16) * EMB;

  for (int ks = 0; ks < KSTEPS; ++ks) {
    const int kb = ks * 32;
    // A fragment: lane = row M, elements 0..15 -> K = kb+half*16+0..15
    v4f a0 = *(const v4f*)(xrow + kb + half * 16);
    v4f a1 = *(const v4f*)(xrow + kb + half * 16 + 4);
    v4f a2 = *(const v4f*)(xrow + kb + half * 16 + 8);
    v4f a3 = *(const v4f*)(xrow + kb + half * 16 + 12);
    v16h a;
#pragma unroll
    for (int i = 0; i < 4; ++i) {
      a[i]      = (_Float16)a0[i];
      a[4 + i]  = (_Float16)a1[i];
      a[8 + i]  = (_Float16)a2[i];
      a[12 + i] = (_Float16)a3[i];
    }
#pragma unroll
    for (int t = 0; t < 4; ++t) {
      // B fragment: B[k][n] = W[n][k]; lane = col N, elements = K kb+half*16+0..15
      const float* wrow = W + (size_t)(nBase + t * 16 + l16) * EMB + kb + half * 16;
      v4f b0 = *(const v4f*)(wrow);
      v4f b1 = *(const v4f*)(wrow + 4);
      v4f b2 = *(const v4f*)(wrow + 8);
      v4f b3 = *(const v4f*)(wrow + 12);
      v16h b;
#pragma unroll
      for (int i = 0; i < 4; ++i) {
        b[i]      = (_Float16)b0[i];
        b[4 + i]  = (_Float16)b1[i];
        b[8 + i]  = (_Float16)b2[i];
        b[12 + i] = (_Float16)b3[i];
      }
      acc[t] = wmma_f16(a, b, acc[t]);
    }
  }

  // Epilogue: C/D layout -> lane l, vgpr j holds D[8*half + j][l16]
#pragma unroll
  for (int t = 0; t < 4; ++t) {
    const int col  = nBase + t * 16 + l16;
    const int hidx = col / HD;
    const int dd   = col % HD;
    const float bv = bias[col];
#pragma unroll
    for (int j = 0; j < 8; ++j) {
      const int r = mBase + half * 8 + j;
      const int n = r / SEQ;
      const int l = r % SEQ;
      const float v = (acc[t][j] + bv) * scale;
      size_t idx;
      if (TRANSPOSED)
        idx = (((size_t)n * HEADS + hidx) * HD + dd) * SEQ + l;
      else
        idx = (((size_t)n * HEADS + hidx) * SEQ + l) * HD + dd;
      out[idx] = (_Float16)v;
    }
  }
}

// ---------------------------------------------------------------------------
// Flash attention. Block = 4 waves, all provably on the same (n,h)
// (128 query-tiles per head, 4 | 128), each wave owns one 16-query tile.
// K/V chunks (32 keys) are staged once per block into double-buffered LDS
// via async global->LDS copies; one barrier per chunk.
// grid.x = BATCH*HEADS*(SEQ/16)/4 = 1024, block = 128.
// ---------------------------------------------------------------------------
__global__ __launch_bounds__(128) void attn_kernel(
    const _Float16* __restrict__ Qh, const _Float16* __restrict__ Kh,
    const _Float16* __restrict__ Vt, _Float16* __restrict__ attOut) {
  __shared__ __align__(16) _Float16 Kb[2][32 * HD];   // 2 x 4 KB, [key][dd]
  __shared__ __align__(16) _Float16 Vb[2][HD * 32];   // 2 x 4 KB, [dd][key]
  __shared__ __align__(16) _Float16 Pl[4][16 * 32];   // 1 KB per wave

  const int tid  = threadIdx.x;
  const int lane = tid & 31;
  const int wave = tid >> 5;
  const int half = lane >> 4;
  const int l16  = lane & 15;

  const int g     = blockIdx.x * 4 + wave;
  const int qt    = g % (SEQ / 16);
  const int nh    = g / (SEQ / 16);
  const int qBase = qt * 16;

  const _Float16* Qp = Qh + (size_t)nh * SEQ * HD;
  const _Float16* Kp = Kh + (size_t)nh * SEQ * HD;
  const _Float16* Vp = Vt + (size_t)nh * HD * SEQ;

  // Cache Q A-fragments for both 32-wide d-chunks for the whole key loop.
  v16h qa[2];
#pragma unroll
  for (int c = 0; c < 2; ++c) {
    const _Float16* qr = Qp + (size_t)(qBase + l16) * HD + c * 32;
    v8h lo = *(const v8h*)(qr + half * 16);
    v8h hi = *(const v8h*)(qr + half * 16 + 8);
#pragma unroll
    for (int i = 0; i < 8; ++i) { qa[c][i] = lo[i]; qa[c][8 + i] = hi[i]; }
  }

  v8f o[4] = {};
  float mrow[8], lrow[8];
#pragma unroll
  for (int j = 0; j < 8; ++j) { mrow[j] = -1e30f; lrow[j] = 0.0f; }

  _Float16* P = &Pl[wave][0];

  // Cooperative stage of one 32-key chunk: K chunk is 4 KB contiguous
  // (32 rows x 64 halfs), V chunk is 64 rows x 64 B. 256 16-byte segments
  // each; 128 threads x 2 segments.
  auto stage_kv = [&](int kc, _Float16* kb, _Float16* vb) {
    const _Float16* kg = Kp + (size_t)kc * HD;
#pragma unroll
    for (int r = 0; r < 2; ++r) {
      const int seg = tid + r * 128;        // 0..255
      copy16_g2l(kb + seg * 8, kg + seg * 8);
      const int row = seg >> 2, col = seg & 3;
      copy16_g2l(vb + row * 32 + col * 8, Vp + (size_t)row * SEQ + kc + col * 8);
    }
  };

  const int NCH = SEQ / 32;  // 64 chunks
  stage_kv(0, Kb[0], Vb[0]);

  for (int i = 0; i < NCH; ++i) {
    const int cur = i & 1;
    wait_async_lds();
    __syncthreads();  // chunk `cur` staged; all waves done reading `1-cur`
    if (i + 1 < NCH) stage_kv((i + 1) * 32, Kb[1 - cur], Vb[1 - cur]);

    const _Float16* kb = &Kb[cur][0];
    const _Float16* vb = &Vb[cur][0];

    // S = Q K^T for 32 keys, as two 16x16 tiles
    v8f s0 = {}, s1 = {};
#pragma unroll
    for (int c = 0; c < 2; ++c) {
      v16h b0 = *(const v16h*)(kb + (size_t)l16 * HD + c * 32 + half * 16);
      v16h b1 = *(const v16h*)(kb + (size_t)(16 + l16) * HD + c * 32 + half * 16);
      s0 = wmma_f16(qa[c], b0, s0);
      s1 = wmma_f16(qa[c], b1, s1);
    }

    // Online softmax. Row m = 8*half + j lives entirely inside one 16-lane
    // half, so the 16-lane DPP butterfly reduces exactly over that row.
    float p0[8], p1[8];
#pragma unroll
    for (int j = 0; j < 8; ++j) {
      const float cm    = redmax16(fmaxf(s0[j], s1[j]));
      const float mnew  = fmaxf(mrow[j], cm);
      const float alpha = __expf(mrow[j] - mnew);
      p0[j] = __expf(s0[j] - mnew);
      p1[j] = __expf(s1[j] - mnew);
      const float rs = redsum16(p0[j] + p1[j]);
      lrow[j] = lrow[j] * alpha + rs;
      mrow[j] = mnew;
#pragma unroll
      for (int t = 0; t < 4; ++t) o[t][j] *= alpha;
    }

    // Relayout P: C/D layout -> row-major 16x32 in LDS -> A fragment.
#pragma unroll
    for (int j = 0; j < 8; ++j) {
      const int row = half * 8 + j;
      P[row * 32 + l16]      = (_Float16)p0[j];
      P[row * 32 + 16 + l16] = (_Float16)p1[j];
    }
    // Intra-wave RAW through LDS: DS ops are in-order per wave; wait for the
    // stores and fence the compiler before the cross-lane reads.
    asm volatile("s_wait_dscnt 0" ::: "memory");

    v8h plo = *(const v8h*)(P + l16 * 32 + half * 16);
    v8h phi = *(const v8h*)(P + l16 * 32 + half * 16 + 8);
    v16h pa;
#pragma unroll
    for (int i2 = 0; i2 < 8; ++i2) { pa[i2] = plo[i2]; pa[8 + i2] = phi[i2]; }

    // O(16x64) += P(16x32) @ Vchunk(32x64); Vb is [dd][key] so B loads are
    // contiguous per lane (lane = output dim).
#pragma unroll
    for (int t = 0; t < 4; ++t) {
      v16h vt = *(const v16h*)(vb + (size_t)(t * 16 + l16) * 32 + half * 16);
      o[t] = wmma_f16(pa, vt, o[t]);
    }
  }

  // Normalize and store f16 att output as [N, L, H*d] for the final GEMM.
  const int n    = nh / HEADS;
  const int hidx = nh % HEADS;
#pragma unroll
  for (int j = 0; j < 8; ++j) {
    const float rinv = 1.0f / lrow[j];
    const int q = qBase + half * 8 + j;
#pragma unroll
    for (int t = 0; t < 4; ++t) {
      const int col = hidx * HD + t * 16 + l16;
      attOut[((size_t)n * SEQ + q) * EMB + col] = (_Float16)(o[t][j] * rinv);
    }
  }
}

// ---------------------------------------------------------------------------
// Output projection: out = att(f16) @ Wo^T + bo, f32 out, row-major.
// grid = (MROWS/64, EMB/64), block = 128.
// ---------------------------------------------------------------------------
__global__ __launch_bounds__(128) void oproj_kernel(
    const _Float16* __restrict__ A, const float* __restrict__ W,
    const float* __restrict__ bias, float* __restrict__ out) {
  const int lane = threadIdx.x & 31;
  const int wave = threadIdx.x >> 5;
  const int half = lane >> 4;
  const int l16  = lane & 15;

  const int mBase = (blockIdx.x * 4 + wave) * 16;
  const int nBase = blockIdx.y * 64;

  v8f acc[4] = {};
  const _Float16* arow = A + (size_t)(mBase + l16) * EMB;

  for (int ks = 0; ks < KSTEPS; ++ks) {
    const int kb = ks * 32;
    v8h alo = *(const v8h*)(arow + kb + half * 16);
    v8h ahi = *(const v8h*)(arow + kb + half * 16 + 8);
    v16h a;
#pragma unroll
    for (int i = 0; i < 8; ++i) { a[i] = alo[i]; a[8 + i] = ahi[i]; }
#pragma unroll
    for (int t = 0; t < 4; ++t) {
      const float* wrow = W + (size_t)(nBase + t * 16 + l16) * EMB + kb + half * 16;
      v4f b0 = *(const v4f*)(wrow);
      v4f b1 = *(const v4f*)(wrow + 4);
      v4f b2 = *(const v4f*)(wrow + 8);
      v4f b3 = *(const v4f*)(wrow + 12);
      v16h b;
#pragma unroll
      for (int i = 0; i < 4; ++i) {
        b[i]      = (_Float16)b0[i];
        b[4 + i]  = (_Float16)b1[i];
        b[8 + i]  = (_Float16)b2[i];
        b[12 + i] = (_Float16)b3[i];
      }
      acc[t] = wmma_f16(a, b, acc[t]);
    }
  }

#pragma unroll
  for (int t = 0; t < 4; ++t) {
    const int col  = nBase + t * 16 + l16;
    const float bv = bias[col];
#pragma unroll
    for (int j = 0; j < 8; ++j) {
      const int r = mBase + half * 8 + j;
      out[(size_t)r * EMB + col] = acc[t][j] + bv;
    }
  }
}

// ---------------------------------------------------------------------------
extern "C" void kernel_launch(void* const* d_in, const int* in_sizes, int n_in,
                              void* d_out, int out_size, void* d_ws, size_t ws_size,
                              hipStream_t stream) {
  (void)in_sizes; (void)n_in; (void)out_size; (void)ws_size;

  const float* q  = (const float*)d_in[0];
  const float* k  = (const float*)d_in[1];
  const float* v  = (const float*)d_in[2];
  const float* Wq = (const float*)d_in[3];
  const float* bq = (const float*)d_in[4];
  const float* Wk = (const float*)d_in[5];
  const float* bk = (const float*)d_in[6];
  const float* Wv = (const float*)d_in[7];
  const float* bv = (const float*)d_in[8];
  const float* Wo = (const float*)d_in[9];
  const float* bo = (const float*)d_in[10];
  float* out = (float*)d_out;

  // Workspace carve-up: 4 f16 tensors of 4096*1024 (8 MB each, 32 MB total).
  _Float16* Qh = (_Float16*)d_ws;
  _Float16* Kh = Qh + (size_t)MROWS * EMB;
  _Float16* Vt = Kh + (size_t)MROWS * EMB;
  _Float16* At = Vt + (size_t)MROWS * EMB;

  dim3 tb(128);
  dim3 gb(MROWS / 64, EMB / 64);          // 64 x 16 blocks
  const float qscale = 0.125f;            // 1/sqrt(HD)

  proj_kernel<0><<<gb, tb, 0, stream>>>(q, Wq, bq, Qh, qscale);
  proj_kernel<0><<<gb, tb, 0, stream>>>(k, Wk, bk, Kh, 1.0f);
  proj_kernel<1><<<gb, tb, 0, stream>>>(v, Wv, bv, Vt, 1.0f);

  dim3 ga(BATCH * HEADS * (SEQ / 16) / 4);  // 1024 blocks of 4 waves
  attn_kernel<<<ga, tb, 0, stream>>>(Qh, Kh, Vt, At);

  oproj_kernel<<<gb, tb, 0, stream>>>(At, Wo, bo, out);
}
